// EfficientAttention_34445637714059
// MI455X (gfx1250) — matrix operand order repaired and checked by reading, not compiled, not run on hardware
//
#include <hip/hip_runtime.h>

// ---------------------------------------------------------------------------
// Linear attention (B=8, S=4096, D=Dv=1024), fp32 in/out.
//   KV[b]  = (K[b] * 1/64)^T @ V[b]        (M=1024, N=1024, Kd=4096)
//   Out[b] = (Q[b] * 1/64)  @ KV[b]        (M=4096, N=1024, Kd=1024)
// bf16 multiplicands + f32 WMMA accumulation on the CDNA5 matrix pipe.
// ---------------------------------------------------------------------------

typedef __attribute__((ext_vector_type(16))) __bf16 v16bf;
typedef __attribute__((ext_vector_type(8)))  __bf16 v8bf;
typedef __attribute__((ext_vector_type(4)))  __bf16 v4bf;
typedef __attribute__((ext_vector_type(8)))  float  v8f;

#define BM 128
#define BN 128
#define BK 32
#define LDK 40   // padded LDS row stride in bf16 elements (80 B, 16B-aligned rows)

static constexpr int BATCH = 8;
static constexpr int SEQ   = 4096;
static constexpr int DIM   = 1024;
static constexpr float QK_SCALE = 1.0f / 64.0f;  // 1/sqrt(4096), exact in bf16

// Assemble a v16bf WMMA fragment from two contiguous 16-byte LDS chunks.
__device__ __forceinline__ v16bf pack_frag(const __bf16* p0, const __bf16* p1) {
    v8bf lo = *(const v8bf*)p0;
    v8bf hi = *(const v8bf*)p1;
    return __builtin_shufflevector(lo, hi, 0, 1, 2, 3, 4, 5, 6, 7,
                                           8, 9, 10, 11, 12, 13, 14, 15);
}

// ---------------------------------------------------------------------------
// Pass 1: KV[b][d][e] = sum_s (K[b][s][d] * 1/64) * V[b][s][e]   (bf16 out)
// grid = (DIM/BM, DIM/BN, BATCH), block = 256
// ---------------------------------------------------------------------------
__global__ __launch_bounds__(256)
void linattn_kv_kernel(const float* __restrict__ Kp,
                       const float* __restrict__ Vp,
                       __bf16* __restrict__ KVp) {
    __shared__ __bf16 As[BM * LDK];  // As[m][kk] = K[s0+kk][m0+m] * scale
    __shared__ __bf16 Bs[BN * LDK];  // Bs[n][kk] = V[s0+kk][n0+n]

    const int tid  = threadIdx.x;
    const int lane = tid & 31;
    const int wave = tid >> 5;
    const int wm   = wave & 3;   // 4 wave-rows of 32
    const int wn   = wave >> 2;  // 2 wave-cols of 64
    const int lrow = lane & 15;
    const int lhi  = lane >> 4;  // 0: lanes 0-15, 1: lanes 16-31

    const int m0 = blockIdx.x * BM;  // d
    const int n0 = blockIdx.y * BN;  // e
    const int b  = blockIdx.z;

    const float* Kb  = Kp + (size_t)b * SEQ * DIM;
    const float* Vb  = Vp + (size_t)b * SEQ * DIM;
    __bf16*      KVb = KVp + (size_t)b * DIM * DIM;

    v8f c[2][4] = {};

    for (int s0 = 0; s0 < SEQ; s0 += BK) {
        // Stage: 32 K-rows x 128 cols of both K^T and V, fp32 -> bf16,
        // transposed into K-contiguous LDS layout.
#pragma unroll
        for (int i = 0; i < 4; ++i) {
            int L  = i * 256 + tid;       // 0..1023
            int kk = L >> 5;              // 0..31  (s offset)
            int m4 = (L & 31) << 2;       // 0..124 (col offset, float4)
            float4 f = *(const float4*)(Kb + (size_t)(s0 + kk) * DIM + m0 + m4);
            As[(m4 + 0) * LDK + kk] = (__bf16)(f.x * QK_SCALE);
            As[(m4 + 1) * LDK + kk] = (__bf16)(f.y * QK_SCALE);
            As[(m4 + 2) * LDK + kk] = (__bf16)(f.z * QK_SCALE);
            As[(m4 + 3) * LDK + kk] = (__bf16)(f.w * QK_SCALE);
            float4 g = *(const float4*)(Vb + (size_t)(s0 + kk) * DIM + n0 + m4);
            Bs[(m4 + 0) * LDK + kk] = (__bf16)g.x;
            Bs[(m4 + 1) * LDK + kk] = (__bf16)g.y;
            Bs[(m4 + 2) * LDK + kk] = (__bf16)g.z;
            Bs[(m4 + 3) * LDK + kk] = (__bf16)g.w;
        }
        __syncthreads();

        // A fragments (16x32): lanes 0-15 hold K[0..8)+K[16..24), 16-31 the rest
        v16bf af[2];
#pragma unroll
        for (int i = 0; i < 2; ++i) {
            int row = wm * 32 + i * 16 + lrow;
            int kb  = lhi * 8;
            af[i] = pack_frag(&As[row * LDK + kb], &As[row * LDK + kb + 16]);
        }
        // B fragments (32x16): lanes 0-15 hold K[0..8)+K[16..24), 16-31 the rest
        v16bf bfr[4];
#pragma unroll
        for (int j = 0; j < 4; ++j) {
            int col = wn * 64 + j * 16 + lrow;
            int kb  = lhi * 8;
            bfr[j] = pack_frag(&Bs[col * LDK + kb], &Bs[col * LDK + kb + 16]);
        }
#pragma unroll
        for (int i = 0; i < 2; ++i)
#pragma unroll
            for (int j = 0; j < 4; ++j)
                c[i][j] = __builtin_amdgcn_wmma_f32_16x16x32_bf16(
                    false, af[i], false, bfr[j], (short)0, c[i][j], false, false);
        __syncthreads();
    }

    // Store C (16x16 f32 layout: vgpr r, lane l -> m = r + (l>>4)*8, n = l&15)
#pragma unroll
    for (int i = 0; i < 2; ++i) {
#pragma unroll
        for (int j = 0; j < 4; ++j) {
            int gm = m0 + wm * 32 + i * 16 + lhi * 8;
            int gn = n0 + wn * 64 + j * 16 + lrow;
#pragma unroll
            for (int r = 0; r < 8; ++r)
                KVb[(size_t)(gm + r) * DIM + gn] = (__bf16)c[i][j][r];
        }
    }
}

// ---------------------------------------------------------------------------
// Pass 2: Out[b][s][e] = sum_d (Q[b][s][d] * 1/64) * KV[b][d][e]   (f32 out)
// grid = (SEQ/BM, DIM/BN, BATCH), block = 256
// ---------------------------------------------------------------------------
__global__ __launch_bounds__(256)
void linattn_out_kernel(const float* __restrict__ Qp,
                        const __bf16* __restrict__ KVp,
                        float* __restrict__ Op) {
    __shared__ __bf16 As[BM * LDK];  // As[m][kk] = Q[m0+m][d0+kk] * scale
    __shared__ __bf16 Bs[BN * LDK];  // Bs[n][kk] = KV[d0+kk][n0+n]

    const int tid  = threadIdx.x;
    const int lane = tid & 31;
    const int wave = tid >> 5;
    const int wm   = wave & 3;
    const int wn   = wave >> 2;
    const int lrow = lane & 15;
    const int lhi  = lane >> 4;

    const int m0 = blockIdx.x * BM;  // s
    const int n0 = blockIdx.y * BN;  // e
    const int b  = blockIdx.z;

    const float*  Qb  = Qp + (size_t)b * SEQ * DIM;
    const __bf16* KVb = KVp + (size_t)b * DIM * DIM;
    float*        Ob  = Op + (size_t)b * SEQ * DIM;

    v8f c[2][4] = {};

    for (int d0 = 0; d0 < DIM; d0 += BK) {
        // A: Q rows are already K-contiguous -> direct b64 bf16 stores
#pragma unroll
        for (int i = 0; i < 4; ++i) {
            int L  = i * 256 + tid;   // 0..1023
            int m  = L >> 3;          // 0..127
            int k4 = (L & 7) << 2;    // 0..28
            float4 f = *(const float4*)(Qb + (size_t)(m0 + m) * DIM + d0 + k4);
            v4bf h;
            h[0] = (__bf16)(f.x * QK_SCALE);
            h[1] = (__bf16)(f.y * QK_SCALE);
            h[2] = (__bf16)(f.z * QK_SCALE);
            h[3] = (__bf16)(f.w * QK_SCALE);
            *(v4bf*)&As[m * LDK + k4] = h;
        }
        // B: KV is bf16 row-major [d][e]; transpose into K-contiguous Bs
#pragma unroll
        for (int i = 0; i < 2; ++i) {
            int L  = i * 256 + tid;   // 0..511
            int kk = L >> 4;          // 0..31 (d offset)
            int n8 = (L & 15) << 3;   // 0..120 (e offset, 8 at a time)
            v8bf h = *(const v8bf*)(KVb + (size_t)(d0 + kk) * DIM + n0 + n8);
#pragma unroll
            for (int j = 0; j < 8; ++j)
                Bs[(n8 + j) * LDK + kk] = h[j];
        }
        __syncthreads();

        v16bf af[2];
#pragma unroll
        for (int i = 0; i < 2; ++i) {
            int row = wm * 32 + i * 16 + lrow;
            int kb  = lhi * 8;
            af[i] = pack_frag(&As[row * LDK + kb], &As[row * LDK + kb + 16]);
        }
        v16bf bfr[4];
#pragma unroll
        for (int j = 0; j < 4; ++j) {
            int col = wn * 64 + j * 16 + lrow;
            int kb  = lhi * 8;
            bfr[j] = pack_frag(&Bs[col * LDK + kb], &Bs[col * LDK + kb + 16]);
        }
#pragma unroll
        for (int i = 0; i < 2; ++i)
#pragma unroll
            for (int j = 0; j < 4; ++j)
                c[i][j] = __builtin_amdgcn_wmma_f32_16x16x32_bf16(
                    false, af[i], false, bfr[j], (short)0, c[i][j], false, false);
        __syncthreads();
    }

#pragma unroll
    for (int i = 0; i < 2; ++i) {
#pragma unroll
        for (int j = 0; j < 4; ++j) {
            int gm = m0 + wm * 32 + i * 16 + lhi * 8;
            int gn = n0 + wn * 64 + j * 16 + lrow;
#pragma unroll
            for (int r = 0; r < 8; ++r)
                Ob[(size_t)(gm + r) * DIM + gn] = c[i][j][r];
        }
    }
}

// ---------------------------------------------------------------------------
extern "C" void kernel_launch(void* const* d_in, const int* in_sizes, int n_in,
                              void* d_out, int out_size, void* d_ws, size_t ws_size,
                              hipStream_t stream) {
    const float* q = (const float*)d_in[0];
    const float* k = (const float*)d_in[1];
    const float* v = (const float*)d_in[2];
    float* out = (float*)d_out;
    __bf16* kv = (__bf16*)d_ws;  // BATCH * DIM * DIM bf16 = 16 MiB scratch

    dim3 grid1(DIM / BM, DIM / BN, BATCH);   // 8 x 8 x 8
    dim3 grid2(SEQ / BM, DIM / BN, BATCH);   // 32 x 8 x 8
    linattn_kv_kernel<<<grid1, 256, 0, stream>>>(k, v, kv);
    linattn_out_kernel<<<grid2, 256, 0, stream>>>(q, kv, out);
}
